// InrAwaSA_59107339927929
// MI455X (gfx1250) — compile-verified, matrix operand order checked by reading, not run on hardware
//
#include <hip/hip_runtime.h>
#include <hip/hip_bf16.h>
#include <math.h>

// Problem constants (from reference): B=4, S=2048, D=1024
#define BB 4
#define SS 2048
#define DD 1024

typedef __attribute__((ext_vector_type(16))) __bf16 v16bf;
typedef __attribute__((ext_vector_type(8)))  __bf16 v8bf;
typedef __attribute__((ext_vector_type(8)))  float  v8f;
typedef __attribute__((ext_vector_type(4)))  float  f4;
typedef __attribute__((ext_vector_type(4)))  unsigned int u4;

__device__ __forceinline__ unsigned short f32_to_bf16(float f) {
    union { float f; unsigned u; } v; v.f = f;
    unsigned r = v.u + 0x7FFFu + ((v.u >> 16) & 1u);  // round-to-nearest-even
    return (unsigned short)(r >> 16);
}

__device__ __forceinline__ unsigned pack_bf16x2(float a, float b) {
    return (unsigned)f32_to_bf16(a) | ((unsigned)f32_to_bf16(b) << 16);
}

// ---------------------------------------------------------------------------
// fp32 -> bf16 conversion, 8 elements per thread, 128-bit stores.
// ---------------------------------------------------------------------------
__global__ void convert_f32_bf16_kernel(const float* __restrict__ src,
                                        unsigned int* __restrict__ dst, int n8) {
    int i = blockIdx.x * blockDim.x + threadIdx.x;   // index in units of 8 floats
    if (i >= n8) return;
    f4 a = reinterpret_cast<const f4*>(src)[2 * i + 0];
    f4 b = reinterpret_cast<const f4*>(src)[2 * i + 1];
    u4 o;
    o.x = pack_bf16x2(a.x, a.y);
    o.y = pack_bf16x2(a.z, a.w);
    o.z = pack_bf16x2(b.x, b.y);
    o.w = pack_bf16x2(b.z, b.w);
    reinterpret_cast<u4*>(dst)[i] = o;
}

// ---------------------------------------------------------------------------
// bf16 fragment load: two 16-byte chunks (K 0..7 / 16..23 for lanes 0-15,
// K 8..15 / 24..31 for lanes 16-31), matching the 16-bit A/B VGPR layout.
// ---------------------------------------------------------------------------
__device__ __forceinline__ v16bf load_frag(const unsigned short* __restrict__ p) {
    v8bf lo = *reinterpret_cast<const v8bf*>(p);
    v8bf hi = *reinterpret_cast<const v8bf*>(p + 16);
    return __builtin_shufflevector(lo, hi, 0,1,2,3,4,5,6,7,8,9,10,11,12,13,14,15);
}

// ---------------------------------------------------------------------------
// Register-blocked wave GEMM: one wave32 computes a 64x64 fp32 block
// (4 M-tiles x 4 N-tiles) with v_wmma_f32_16x16x32_bf16.
// Per k-step: 8 fragment loads (16 x b128) feed 16 WMMAs -> 1 load/WMMA.
// LDA/LDB are compile-time so every fragment address is one of TWO base
// pointers + a constant instruction offset (i*16*LDA fits 24-bit IOFFSET).
// ---------------------------------------------------------------------------
template <int LDA, int LDB>
__device__ __forceinline__ void wmma_block(const unsigned short* __restrict__ A,
                                           const unsigned short* __restrict__ Bm,
                                           int kTotal, int aRow0, int bCol0,
                                           int lm, int lh, v8f c[4][4]) {
    const unsigned short* aBase = A + (size_t)(aRow0 + lm) * LDA + lh * 8;
    const unsigned short* bBase = Bm + (size_t)(bCol0 + lm) * LDB + lh * 8;

#pragma unroll
    for (int i = 0; i < 4; ++i)
#pragma unroll
        for (int j = 0; j < 4; ++j)
            c[i][j] = (v8f){};

    for (int k = 0; k < kTotal; k += 32) {
        v16bf a[4], b[4];
#pragma unroll
        for (int i = 0; i < 4; ++i) a[i] = load_frag(aBase + k + i * 16 * LDA);
#pragma unroll
        for (int j = 0; j < 4; ++j) b[j] = load_frag(bBase + k + j * 16 * LDB);
#pragma unroll
        for (int i = 0; i < 4; ++i)
#pragma unroll
            for (int j = 0; j < 4; ++j)
                c[i][j] = __builtin_amdgcn_wmma_f32_16x16x32_bf16(
                    false, a[i], false, b[j], (short)0, c[i][j], false, false);
    }
}

// Wave placement inside a 256-thread block: 8 waves tile 128(M) x 256(N):
//   waveM = wave >> 2 (2 waves x 64 rows), waveN = wave & 3 (4 waves x 64 cols).
#define WAVE_GEOM()                                        \
    const int wave = threadIdx.x >> 5;                     \
    const int lane = threadIdx.x & 31;                     \
    const int lm = lane & 15, lh = lane >> 4;              \
    const int mW = (wave >> 2) * 64;                       \
    const int nW = (wave & 3) * 64;

// ---------------------------------------------------------------------------
// Projections: out[m,n] = sum_d x[m,d] * W[n,d]
// grid = (D/256, B*S/128). transposeStore=1 writes [B][D][S] (for V).
// ---------------------------------------------------------------------------
__global__ void __launch_bounds__(256, 1)
gemm_proj_kernel(const unsigned short* __restrict__ xb,
                 const unsigned short* __restrict__ wb,
                 unsigned short* __restrict__ outb,
                 float scale, int transposeStore) {
    WAVE_GEOM();
    const int mBase = blockIdx.y * 128 + mW;
    const int nBase = blockIdx.x * 256 + nW;

    v8f c[4][4];
    wmma_block<DD, DD>(xb, wb, DD, mBase, nBase, lm, lh, c);

    if (!transposeStore) {
#pragma unroll
        for (int i = 0; i < 4; ++i)
#pragma unroll
            for (int j = 0; j < 4; ++j) {
                int col = nBase + j * 16 + lm;
#pragma unroll
                for (int r = 0; r < 8; ++r) {
                    int row = mBase + i * 16 + r + 8 * lh;
                    outb[(size_t)row * DD + col] = f32_to_bf16(c[i][j][r] * scale);
                }
            }
    } else {
        const int b = mBase / SS;
        const int s0 = mBase % SS;
        unsigned short* ob = outb + (size_t)b * DD * SS;
#pragma unroll
        for (int i = 0; i < 4; ++i)
#pragma unroll
            for (int j = 0; j < 4; ++j) {
                int col = nBase + j * 16 + lm;
#pragma unroll
                for (int r = 0; r < 8; ++r) {
                    int s = s0 + i * 16 + r + 8 * lh;
                    ob[(size_t)col * SS + s] = f32_to_bf16(c[i][j][r] * scale);
                }
            }
    }
}

// ---------------------------------------------------------------------------
// scores[b,q,k] = sum_e Q[b,q,e] * K[b,k,e]   grid = (S/256, S/128, B)
// ---------------------------------------------------------------------------
__global__ void __launch_bounds__(256, 1)
gemm_scores_kernel(const unsigned short* __restrict__ Qb,
                   const unsigned short* __restrict__ Kb,
                   float* __restrict__ scores) {
    WAVE_GEOM();
    const int b = blockIdx.z;
    const int mBase = blockIdx.y * 128 + mW;
    const int nBase = blockIdx.x * 256 + nW;

    v8f c[4][4];
    wmma_block<DD, DD>(Qb + (size_t)b * SS * DD, Kb + (size_t)b * SS * DD,
                       DD, mBase, nBase, lm, lh, c);

    float* out = scores + (size_t)b * SS * SS;
#pragma unroll
    for (int i = 0; i < 4; ++i)
#pragma unroll
        for (int j = 0; j < 4; ++j) {
            int col = nBase + j * 16 + lm;
#pragma unroll
            for (int r = 0; r < 8; ++r) {
                int row = mBase + i * 16 + r + 8 * lh;
                out[(size_t)row * SS + col] = c[i][j][r];
            }
        }
}

// ---------------------------------------------------------------------------
// Row softmax over 2048 fp32 scores -> bf16 probabilities. One block per row.
// Thread t owns the contiguous chunk [t*8, t*8+8): two float4 loads, one
// uint4 (8x bf16) store.
// ---------------------------------------------------------------------------
__global__ void softmax_kernel(const float* __restrict__ scores,
                               unsigned int* __restrict__ prob) {
    const int row = blockIdx.x;                 // 0 .. B*S-1
    const int tid = threadIdx.x;                // 256 threads
    const f4* rp = reinterpret_cast<const f4*>(scores + (size_t)row * SS);
    u4* pp = reinterpret_cast<u4*>(prob + (size_t)row * (SS / 2));

    __shared__ float red[256];

    f4 a = rp[2 * tid + 0];
    f4 b = rp[2 * tid + 1];
    float m = fmaxf(fmaxf(fmaxf(a.x, a.y), fmaxf(a.z, a.w)),
                    fmaxf(fmaxf(b.x, b.y), fmaxf(b.z, b.w)));
    red[tid] = m;
    __syncthreads();
    for (int s = 128; s > 0; s >>= 1) {
        if (tid < s) red[tid] = fmaxf(red[tid], red[tid + s]);
        __syncthreads();
    }
    m = red[0];
    __syncthreads();

    a.x = __expf(a.x - m); a.y = __expf(a.y - m);
    a.z = __expf(a.z - m); a.w = __expf(a.w - m);
    b.x = __expf(b.x - m); b.y = __expf(b.y - m);
    b.z = __expf(b.z - m); b.w = __expf(b.w - m);
    float sum = (a.x + a.y) + (a.z + a.w) + (b.x + b.y) + (b.z + b.w);
    red[tid] = sum;
    __syncthreads();
    for (int s = 128; s > 0; s >>= 1) {
        if (tid < s) red[tid] += red[tid + s];
        __syncthreads();
    }
    const float inv = 1.0f / red[0];

    u4 o;
    o.x = pack_bf16x2(a.x * inv, a.y * inv);
    o.y = pack_bf16x2(a.z * inv, a.w * inv);
    o.z = pack_bf16x2(b.x * inv, b.y * inv);
    o.w = pack_bf16x2(b.z * inv, b.w * inv);
    pp[tid] = o;
}

// ---------------------------------------------------------------------------
// out[b,q,e] = sum_k P[b,q,k] * V[b,k,e]; V stored transposed as Vt[b][e][k].
// grid = (D/256, S/128, B); fp32 output.
// ---------------------------------------------------------------------------
__global__ void __launch_bounds__(256, 1)
gemm_pv_kernel(const unsigned short* __restrict__ prob,
               const unsigned short* __restrict__ Vt,
               float* __restrict__ out) {
    WAVE_GEOM();
    const int b = blockIdx.z;
    const int mBase = blockIdx.y * 128 + mW;
    const int nBase = blockIdx.x * 256 + nW;

    v8f c[4][4];
    wmma_block<SS, SS>(prob + (size_t)b * SS * SS, Vt + (size_t)b * DD * SS,
                       SS, mBase, nBase, lm, lh, c);

    float* ob = out + (size_t)b * SS * DD;
#pragma unroll
    for (int i = 0; i < 4; ++i)
#pragma unroll
        for (int j = 0; j < 4; ++j) {
            int col = nBase + j * 16 + lm;
#pragma unroll
            for (int r = 0; r < 8; ++r) {
                int row = mBase + i * 16 + r + 8 * lh;
                ob[(size_t)row * DD + col] = c[i][j][r];
            }
        }
}

// ---------------------------------------------------------------------------
// Host launch
// ---------------------------------------------------------------------------
extern "C" void kernel_launch(void* const* d_in, const int* in_sizes, int n_in,
                              void* d_out, int out_size, void* d_ws, size_t ws_size,
                              hipStream_t stream) {
    (void)in_sizes; (void)n_in; (void)out_size; (void)ws_size;

    // setup_inputs order: x, attn_mask, pad_mask, wq, wk, wv (masks unused per reference bug)
    const float* x  = (const float*)d_in[0];
    const float* wq = (const float*)d_in[3];
    const float* wk = (const float*)d_in[4];
    const float* wv = (const float*)d_in[5];
    float* out = (float*)d_out;

    const size_t nX = (size_t)BB * SS * DD;     // 8,388,608
    const size_t nW = (size_t)DD * DD;          // 1,048,576

    // Workspace layout (bytes, 256-aligned blocks)
    char* ws = (char*)d_ws;
    size_t off = 0;
    auto alloc = [&](size_t bytes) {
        size_t o = off;
        off += (bytes + 255) & ~(size_t)255;
        return o;
    };
    unsigned short* xb   = (unsigned short*)(ws + alloc(nX * 2));
    unsigned short* wqb  = (unsigned short*)(ws + alloc(nW * 2));
    unsigned short* wkb  = (unsigned short*)(ws + alloc(nW * 2));
    unsigned short* wvb  = (unsigned short*)(ws + alloc(nW * 2));
    unsigned short* Qb   = (unsigned short*)(ws + alloc(nX * 2));
    unsigned short* Kb   = (unsigned short*)(ws + alloc(nX * 2));
    unsigned short* Vtb  = (unsigned short*)(ws + alloc(nX * 2));
    float*          scr  = (float*)        (ws + alloc((size_t)BB * SS * SS * 4));
    unsigned short* prb  = (unsigned short*)(ws + alloc((size_t)BB * SS * SS * 2));

    // 1) fp32 -> bf16 conversions (8 elements per thread)
    convert_f32_bf16_kernel<<<dim3((unsigned)(nX / 8 / 256)), 256, 0, stream>>>(
        x, (unsigned int*)xb, (int)(nX / 8));
    convert_f32_bf16_kernel<<<dim3((unsigned)(nW / 8 / 256)), 256, 0, stream>>>(
        wq, (unsigned int*)wqb, (int)(nW / 8));
    convert_f32_bf16_kernel<<<dim3((unsigned)(nW / 8 / 256)), 256, 0, stream>>>(
        wk, (unsigned int*)wkb, (int)(nW / 8));
    convert_f32_bf16_kernel<<<dim3((unsigned)(nW / 8 / 256)), 256, 0, stream>>>(
        wv, (unsigned int*)wvb, (int)(nW / 8));

    // 2) Projections. Block tile 128(M) x 256(N); grid = (D/256, B*S/128)
    dim3 projGrid(DD / 256, (BB * SS) / 128);             // (4, 64)
    const float qscale = 1.0f / 32.0f;                    // 1/sqrt(D), folded into Q
    gemm_proj_kernel<<<projGrid, 256, 0, stream>>>(xb, wqb, Qb,  qscale, 0);
    gemm_proj_kernel<<<projGrid, 256, 0, stream>>>(xb, wkb, Kb,  1.0f,   0);
    gemm_proj_kernel<<<projGrid, 256, 0, stream>>>(xb, wvb, Vtb, 1.0f,   1);  // V^T

    // 3) scores = Q K^T / sqrt(D)
    dim3 scGrid(SS / 256, SS / 128, BB);                  // (8, 16, 4)
    gemm_scores_kernel<<<scGrid, 256, 0, stream>>>(Qb, Kb, scr);

    // 4) softmax rows -> bf16 prob
    softmax_kernel<<<dim3(BB * SS), 256, 0, stream>>>(scr, (unsigned int*)prb);

    // 5) out = P V
    dim3 pvGrid(DD / 256, SS / 128, BB);                  // (4, 16, 4)
    gemm_pv_kernel<<<pvGrid, 256, 0, stream>>>(prb, Vtb, out);
}
